// GCN_51874615001204
// MI455X (gfx1250) — matrix operand order repaired and checked by reading, not compiled, not run on hardware
//
#include <hip/hip_runtime.h>
#include <hip/hip_bf16.h>

// ---------------------------------------------------------------------------
// Types for CDNA5 WMMA (wave32): v_wmma_f32_16x16x32_f16
// ---------------------------------------------------------------------------
typedef _Float16 v16h __attribute__((ext_vector_type(16)));
typedef _Float16 v8h  __attribute__((ext_vector_type(8)));
typedef float    v8f  __attribute__((ext_vector_type(8)));
typedef int      v4i  __attribute__((ext_vector_type(4)));

union HF16 { v16h v; v8h h[2]; };

#define NN   4096   // nodes
#define DD   16     // embedding dim
#define CC   64     // in channels
#define OO   64     // out channels
#define BB   16     // batch
#define BC   1024   // BB*CC folded columns

// ---------------------------------------------------------------------------
// CDNA5 async global->LDS copy (ASYNCcnt-tracked) with portable fallbacks.
// Builtin prototype (probe-confirmed via diagnostic): 
//   void __builtin_amdgcn_global_load_async_to_lds_b128(as1 v4i*, as3 v4i*, Ii, Ii)
// ---------------------------------------------------------------------------
typedef __attribute__((address_space(1))) const _Float16 glb_h_t;
typedef __attribute__((address_space(3))) _Float16       lds_h_t;
typedef __attribute__((address_space(1))) v4i            as1_v4i;
typedef __attribute__((address_space(3))) v4i            as3_v4i;

__device__ __forceinline__ void async_cp16(_Float16* dst_lds,
                                           const _Float16* src_g) {
#if __has_builtin(__builtin_amdgcn_global_load_async_to_lds_b128)
  const glb_h_t* g1 = (const glb_h_t*)src_g;   // addrspacecast generic -> AS1
  lds_h_t*       l1 = (lds_h_t*)dst_lds;       // addrspacecast generic -> AS3
  __builtin_amdgcn_global_load_async_to_lds_b128(
      (as1_v4i*)g1, (as3_v4i*)l1, 0, 0);
#else
  const unsigned lds_off = (unsigned)(unsigned long long)(lds_h_t*)dst_lds;
  asm volatile("global_load_async_to_lds_b128 %0, %1, off"
               :: "v"(lds_off), "v"(src_g)
               : "memory");
#endif
}

template <int N>
__device__ __forceinline__ void wait_async() {
#if __has_builtin(__builtin_amdgcn_s_wait_asynccnt)
  __builtin_amdgcn_s_wait_asynccnt(N);
#else
  asm volatile("s_wait_asynccnt %0" :: "i"(N) : "memory");
#endif
}

// ---------------------------------------------------------------------------
// Kernel 1: A = softmax(relu(E E^T), axis=1) -> f16, row-major [NN][NN].
// One block per row; whole row lives in registers (16 vals/thread).
// ---------------------------------------------------------------------------
__global__ __launch_bounds__(256) void adj_softmax_kernel(
    const float* __restrict__ E, _Float16* __restrict__ Ah) {
  const int n = blockIdx.x;
  const int t = threadIdx.x;
  __shared__ float en[DD];
  __shared__ float red[256];
  if (t < DD) en[t] = E[n * DD + t];
  __syncthreads();
  float e0[DD];
#pragma unroll
  for (int d = 0; d < DD; ++d) e0[d] = en[d];

  float s[16];
  float mx = 0.0f;  // relu(.) >= 0, diagonal > 0, so 0 is a safe lower bound
#pragma unroll 4
  for (int it = 0; it < 16; ++it) {
    const int m = it * 256 + t;
    const float4* ep = (const float4*)(E + (size_t)m * DD);
    float acc = 0.0f;
#pragma unroll
    for (int q = 0; q < 4; ++q) {
      float4 v = ep[q];
      acc += e0[q * 4 + 0] * v.x + e0[q * 4 + 1] * v.y +
             e0[q * 4 + 2] * v.z + e0[q * 4 + 3] * v.w;
    }
    acc = fmaxf(acc, 0.0f);
    s[it] = acc;
    mx = fmaxf(mx, acc);
  }
  red[t] = mx;
  __syncthreads();
  for (int off = 128; off > 0; off >>= 1) {
    if (t < off) red[t] = fmaxf(red[t], red[t + off]);
    __syncthreads();
  }
  mx = red[0];
  __syncthreads();
  float sum = 0.0f;
#pragma unroll
  for (int it = 0; it < 16; ++it) {
    float e = __expf(s[it] - mx);
    s[it] = e;
    sum += e;
  }
  red[t] = sum;
  __syncthreads();
  for (int off = 128; off > 0; off >>= 1) {
    if (t < off) red[t] += red[t + off];
    __syncthreads();
  }
  const float inv = 1.0f / red[0];
#pragma unroll
  for (int it = 0; it < 16; ++it) {
    Ah[(size_t)n * NN + it * 256 + t] = (_Float16)(s[it] * inv);
  }
}

// ---------------------------------------------------------------------------
// Kernel 2: X[B,N,C] f32 -> Xt[BC][NN] f16 where Xt[b*64+c][n] = X[b][n][c].
// K-contiguous layout: directly the GEMM B-operand source.
// ---------------------------------------------------------------------------
__global__ __launch_bounds__(256) void xpose_kernel(
    const float* __restrict__ X, _Float16* __restrict__ Xt) {
  const int blk = blockIdx.x;   // 0..1023
  const int b   = blk >> 6;
  const int n0  = (blk & 63) * 64;
  const int t   = threadIdx.x;
  __shared__ _Float16 tile[64 * 72];
#pragma unroll
  for (int it = 0; it < 16; ++it) {
    const int idx = it * 256 + t;
    const int nl = idx >> 6, c = idx & 63;
    const float v = X[((size_t)b * NN + n0 + nl) * CC + c];
    tile[c * 72 + nl] = (_Float16)v;
  }
  __syncthreads();
#pragma unroll
  for (int it = 0; it < 16; ++it) {
    const int idx = it * 256 + t;
    const int c = idx >> 6, nl = idx & 63;
    Xt[(size_t)(b * 64 + c) * NN + n0 + nl] = tile[c * 72 + nl];
  }
}

// ---------------------------------------------------------------------------
// Kernel 3: WMMA GEMM  C = A * B.
//   A : [NN][NN] f16 row-major (K contiguous)
//   Bt: [BC][NN] f16 (K contiguous per output column)
//   Cf : optional f32 row-major [NN][BC]   (for the fused final stage)
//   CtH: optional f16 transposed [BC][NN]  (B operand of next GEMM)
// Block tile 128x128, 8 waves as 4x2 grid of 32x64 wave tiles, K-step 32.
// Double-buffered LDS fed by GLOBAL_LOAD_ASYNC_TO_LDS_B128 (ASYNCcnt),
// next tile's copies overlap the current tile's WMMAs.
// ---------------------------------------------------------------------------
__global__ __launch_bounds__(256) void gemm_wmma_kernel(
    const _Float16* __restrict__ A, const _Float16* __restrict__ Bt,
    float* __restrict__ Cf, _Float16* __restrict__ CtH,
    int writeF, int writeH) {
  const int ct0 = blockIdx.x * 128;  // column base
  const int rt0 = blockIdx.y * 128;  // row base
  const int t    = threadIdx.x;
  const int wid  = t >> 5;
  const int lane = t & 31;
  const int m16  = lane & 15;
  const int kh   = lane >> 4;
  const int rg   = wid >> 1;   // 0..3: 32-row group
  const int cg   = wid & 1;    // 0..1: 64-col group

  __shared__ _Float16 As[2][128 * 32];  // [buf][row][k]
  __shared__ _Float16 Bs[2][128 * 32];  // [buf][col][k]

  auto issue_tile = [&](int k0, int p) {
#pragma unroll
    for (int it = 0; it < 2; ++it) {
      const int chunk = it * 256 + t;            // 0..511
      const int row = chunk >> 2, kc = (chunk & 3) * 8;
      async_cp16(&As[p][row * 32 + kc],
                 &A[(size_t)(rt0 + row) * NN + k0 + kc]);
      async_cp16(&Bs[p][row * 32 + kc],
                 &Bt[(size_t)(ct0 + row) * NN + k0 + kc]);
    }
  };

  v8f acc[2][4] = {};

  issue_tile(0, 0);
  int p = 0;
  for (int k0 = 0; k0 < NN; k0 += 32) {
    const bool more = (k0 + 32) < NN;
    if (more) {
      issue_tile(k0 + 32, p ^ 1);  // overlap next tile with current WMMAs
      wait_async<4>();             // first 4 (current tile) done, in order
    } else {
      wait_async<0>();
    }
    __syncthreads();

    // A fragments: row m16(+16), two 8-element runs at K = 8*kh and 16+8*kh
    HF16 a0, a1;
    const _Float16* abase = &As[p][(rg * 32 + m16) * 32];
    a0.h[0] = *(const v8h*)(abase + kh * 8);
    a0.h[1] = *(const v8h*)(abase + 16 + kh * 8);
    a1.h[0] = *(const v8h*)(abase + 16 * 32 + kh * 8);
    a1.h[1] = *(const v8h*)(abase + 16 * 32 + 16 + kh * 8);
#pragma unroll
    for (int ct = 0; ct < 4; ++ct) {
      // B fragment: col m16, two 8-element runs at K = 8*kh and 16+8*kh
      HF16 b;
      const _Float16* bpp = &Bs[p][(cg * 64 + ct * 16 + m16) * 32];
      b.h[0] = *(const v8h*)(bpp + kh * 8);
      b.h[1] = *(const v8h*)(bpp + 16 + kh * 8);
      acc[0][ct] = __builtin_amdgcn_wmma_f32_16x16x32_f16(
          false, a0.v, false, b.v, (short)0, acc[0][ct], false, false);
      acc[1][ct] = __builtin_amdgcn_wmma_f32_16x16x32_f16(
          false, a1.v, false, b.v, (short)0, acc[1][ct], false, false);
    }
    __syncthreads();  // buffer p free before it is refilled next-next iter
    p ^= 1;
  }

  // Epilogue. D layout: VGPR r -> row (r + 8*kh), col (lane&15).
#pragma unroll
  for (int rr = 0; rr < 2; ++rr) {
    const int grow0 = rt0 + rg * 32 + rr * 16 + kh * 8;
#pragma unroll
    for (int ct = 0; ct < 4; ++ct) {
      const int gcol = ct0 + cg * 64 + ct * 16 + m16;
      if (writeF) {
#pragma unroll
        for (int r = 0; r < 8; ++r)
          Cf[(size_t)(grow0 + r) * BC + gcol] = acc[rr][ct][r];
      }
      if (writeH) {
        v8h hv;
#pragma unroll
        for (int r = 0; r < 8; ++r) hv[r] = (_Float16)acc[rr][ct][r];
        *(v8h*)&CtH[(size_t)gcol * NN + grow0] = hv;  // packed 16B store
      }
    }
  }
}

// ---------------------------------------------------------------------------
// Kernel 4: fused per-node stage. One block per node n:
//   Wn[o][k*64+i] = sum_d E[n,d] * wp[d,k,i,o]      (f16 B operand in LDS)
//   Xg[b][k*64+i] = {X, Y1, 2*Y2 - X}[b,n,i]        (f16 A operand in LDS)
//   out[b,n,o]    = WMMA([16x192] @ [192x64]) + E[n,:]*bias_pool[:,o]
// ---------------------------------------------------------------------------
__global__ __launch_bounds__(256) void final_node_kernel(
    const float* __restrict__ X,    // [16][4096][64]
    const float* __restrict__ Y1f,  // [4096][1024]
    const float* __restrict__ Y2f,  // [4096][1024]
    const float* __restrict__ E,    // [4096][16]
    const float* __restrict__ wp,   // [16][3][64][64]
    const float* __restrict__ bpool,// [16][64]
    float* __restrict__ out) {      // [16][4096][64]
  const int n    = blockIdx.x;
  const int t    = threadIdx.x;
  const int wid  = t >> 5;
  const int lane = t & 31;
  const int m16  = lane & 15;
  const int kh   = lane >> 4;

  __shared__ _Float16 Wt[64 * 192];  // Wt[o][ki], ki contiguous
  __shared__ _Float16 Xg[16 * 192];  // Xg[b][ki], ki contiguous
  __shared__ float es[DD];
  if (t < DD) es[t] = E[n * DD + t];
  __syncthreads();
  float e0[DD];
#pragma unroll
  for (int d = 0; d < DD; ++d) e0[d] = es[d];

  // Node-adaptive weights: 12288 elements, coalesced over (ki*64 + o).
#pragma unroll 2
  for (int j = 0; j < 48; ++j) {
    const int idx = j * 256 + t;  // = ki*64 + o
    float acc = 0.0f;
#pragma unroll
    for (int d = 0; d < DD; ++d) acc += e0[d] * wp[d * 12288 + idx];
    const int ki = idx >> 6, o = idx & 63;
    Wt[o * 192 + ki] = (_Float16)acc;
  }

  // Chebyshev-gathered features for this node.
#pragma unroll
  for (int j = 0; j < 12; ++j) {
    const int idx = j * 256 + t;  // = b*192 + ki
    const int b = idx / 192, ki = idx % 192;
    const int k = ki >> 6, i = ki & 63;
    float v;
    if (k == 1) {
      v = Y1f[(size_t)n * BC + b * 64 + i];
    } else {
      const float xv = X[((size_t)b * NN + n) * CC + i];
      v = (k == 0) ? xv : (2.0f * Y2f[(size_t)n * BC + b * 64 + i] - xv);
    }
    Xg[b * 192 + ki] = (_Float16)v;
  }
  __syncthreads();

  if (wid < 4) {  // 4 waves cover O=64 (16 cols each); M=16 = batch
    v8f acc = {};
#pragma unroll
    for (int step = 0; step < 6; ++step) {
      const int k0 = step * 32;
      HF16 a, b;
      a.h[0] = *(const v8h*)&Xg[m16 * 192 + k0 + kh * 8];
      a.h[1] = *(const v8h*)&Xg[m16 * 192 + k0 + 16 + kh * 8];
      const _Float16* wtp = &Wt[(wid * 16 + m16) * 192 + k0];
      b.h[0] = *(const v8h*)(wtp + kh * 8);
      b.h[1] = *(const v8h*)(wtp + 16 + kh * 8);
      acc = __builtin_amdgcn_wmma_f32_16x16x32_f16(
          false, a.v, false, b.v, (short)0, acc, false, false);
    }
    const int o = wid * 16 + m16;
    float bias = 0.0f;
#pragma unroll
    for (int d = 0; d < DD; ++d) bias += e0[d] * bpool[d * OO + o];
#pragma unroll
    for (int r = 0; r < 8; ++r) {
      const int b = kh * 8 + r;  // batch index from D layout
      out[((size_t)b * NN + n) * OO + o] = acc[r] + bias;
    }
  }
}

// ---------------------------------------------------------------------------
// Launch: inputs are (X, E, E_t, weights_pool, bias_pool); output f32 [16,4096,64].
// Workspace layout (80 MB total): Ah | Xt | Y1f | Y1t | Y2f
// ---------------------------------------------------------------------------
extern "C" void kernel_launch(void* const* d_in, const int* in_sizes, int n_in,
                              void* d_out, int out_size, void* d_ws, size_t ws_size,
                              hipStream_t stream) {
  (void)in_sizes; (void)n_in; (void)out_size; (void)ws_size;
  const float* X     = (const float*)d_in[0];
  const float* E     = (const float*)d_in[1];
  const float* wp    = (const float*)d_in[3];
  const float* bpool = (const float*)d_in[4];
  float* out = (float*)d_out;

  char* ws = (char*)d_ws;
  _Float16* Ah  = (_Float16*)ws; ws += (size_t)NN * NN * sizeof(_Float16);
  _Float16* Xt  = (_Float16*)ws; ws += (size_t)BC * NN * sizeof(_Float16);
  float*    Y1f = (float*)ws;    ws += (size_t)NN * BC * sizeof(float);
  _Float16* Y1t = (_Float16*)ws; ws += (size_t)BC * NN * sizeof(_Float16);
  float*    Y2f = (float*)ws;    ws += (size_t)NN * BC * sizeof(float);

  adj_softmax_kernel<<<NN, 256, 0, stream>>>(E, Ah);
  xpose_kernel<<<BC, 256, 0, stream>>>(X, Xt);
  dim3 g(BC / 128, NN / 128);
  gemm_wmma_kernel<<<g, 256, 0, stream>>>(Ah, Xt, Y1f, Y1t, 1, 1);
  gemm_wmma_kernel<<<g, 256, 0, stream>>>(Ah, Y1t, Y2f, (_Float16*)nullptr, 1, 0);
  final_node_kernel<<<NN, 256, 0, stream>>>(X, Y1f, Y2f, E, wp, bpool, out);
}
